// Mamba_4664334483707
// MI455X (gfx1250) — matrix operand order repaired and checked by reading, not compiled, not run on hardware
//
#include <hip/hip_runtime.h>
#include <hip/hip_bf16.h>

// ---------------- model config (fixed by reference) ----------------
#define BB      2
#define LL      1024
#define DD      768
#define EE      1536
#define NN      16
#define NDELTA  48
#define NDELTAP 64        // padded to multiple of 32 for WMMA K
#define DCONV   4
#define VV      50280
#define NLAYERS 2
#define MM      (BB*LL)   // 2048 rows

typedef __attribute__((ext_vector_type(16))) _Float16 v16h;
typedef __attribute__((ext_vector_type(8)))  _Float16 v8h;
typedef __attribute__((ext_vector_type(8)))  float    v8f;
typedef __attribute__((ext_vector_type(4)))  unsigned int u32x4;
typedef __attribute__((ext_vector_type(4)))  int      i32x4;
typedef __attribute__((ext_vector_type(8)))  int      i32x8;

// ---------------- helpers ----------------
__device__ __forceinline__ float silu_f(float v) {
    return v / (1.0f + __expf(-v));
}
__device__ __forceinline__ float softplus_f(float v) {
    return (v > 20.0f) ? v : log1pf(__expf(v));
}

// ---------------- embedding gather ----------------
__global__ __launch_bounds__(256)
void k_embed(const int* __restrict__ tokens, const float* __restrict__ emb,
             float* __restrict__ resid) {
    size_t idx = (size_t)blockIdx.x * blockDim.x + threadIdx.x;
    if (idx >= (size_t)MM * DD) return;
    int row = (int)(idx / DD);
    int col = (int)(idx % DD);
    int tok = tokens[row];
    resid[idx] = emb[(size_t)tok * DD + col];
}

// ---------------- RMSNorm (f32 in -> f16 out) ----------------
__global__ __launch_bounds__(256)
void k_rmsnorm(const float* __restrict__ x, const float* __restrict__ w,
               _Float16* __restrict__ out16, int Dd) {
    __shared__ float red[256];
    int row = blockIdx.x;
    int tid = threadIdx.x;
    const float* xr = x + (size_t)row * Dd;
    float s = 0.f;
    for (int i = tid; i < Dd; i += 256) { float v = xr[i]; s += v * v; }
    red[tid] = s;
    __syncthreads();
    for (int st = 128; st > 0; st >>= 1) {
        if (tid < st) red[tid] += red[tid + st];
        __syncthreads();
    }
    float rinv = rsqrtf(red[0] / (float)Dd + 1e-5f);
    _Float16* orow = out16 + (size_t)row * Dd;
    for (int i = tid; i < Dd; i += 256)
        orow[i] = (_Float16)(xr[i] * rinv * w[i]);
}

// ---------------- f32 -> f16 convert with optional column pad ----------------
__global__ __launch_bounds__(256)
void k_cvt_pad(const float* __restrict__ src, _Float16* __restrict__ dst,
               int rows, int cols, int colsPad) {
    size_t idx = (size_t)blockIdx.x * blockDim.x + threadIdx.x;
    size_t total = (size_t)rows * colsPad;
    if (idx >= total) return;
    int c = (int)(idx % colsPad);
    int r = (int)(idx / colsPad);
    dst[idx] = (c < cols) ? (_Float16)src[(size_t)r * cols + c] : (_Float16)0.0f;
}

// ================= WMMA GEMM with TDM double-buffered LDS staging ==========
// C[M,N] = A[M,K] * W[N,K]^T  (+bias, softplus if act==1, +addsrc)
// block: 256 thr = 8 waves; block tile 128(M) x 128(N); K chunk = 32
// wave grid 4(M-groups of 32) x 2(N-groups of 64); 8 wmma / wave / chunk
#define BM 128
#define BN 128
#define BK 32
#define LDST 40                    // padded LDS row stride in f16 (80 B)
#define TILE_HALFS (BM * LDST)     // 5120 halfs = 10240 B per tile buffer

// Issue one TDM 2D tile load: tile BK x 128 rows of f16 from row-major
// [rows x K] matrix into LDS at lds_off, padding each 16-DWORD row with
// 4 DWORDs so LDS row stride is LDST halfs.
__device__ __forceinline__ void tdm_tile_load(unsigned lds_off,
                                              const _Float16* gbase,
                                              unsigned long long elemOff,
                                              int K, int rowsRemain) {
    unsigned long long ga = (unsigned long long)gbase + 2ull * elemOff;
    unsigned td0 = (unsigned)K;                       // tensor dim0 (elems)
    unsigned td1 = (unsigned)(rowsRemain > 0 ? rowsRemain : 0);
    unsigned long long st = (unsigned long long)(unsigned)K;  // dim0 stride

    u32x4 g0;
    g0[0] = 1u;                                       // count=1, normal mode
    g0[1] = lds_off;                                  // LDS byte address
    g0[2] = (unsigned)(ga & 0xFFFFFFFFull);
    g0[3] = (unsigned)((ga >> 32) & 0x1FFFFFFull) | (2u << 30);  // type=2

    i32x8 g1;
    // data_size=1(2B) | pad_enable | pad_interval=16dw(code3) | pad_amount=4dw(code3)
    g1[0] = (int)((1u << 16) | (1u << 20) | (3u << 22) | (3u << 25));
    g1[1] = (int)((td0 & 0xFFFFu) << 16);             // tensor_dim0 lo16
    g1[2] = (int)(((td0 >> 16) & 0xFFFFu) | ((td1 & 0xFFFFu) << 16));
    g1[3] = (int)(((td1 >> 16) & 0xFFFFu) | ((unsigned)BK << 16)); // tile_dim0
    g1[4] = (int)(unsigned)BM;                        // tile_dim1=128, tile_dim2=0
    g1[5] = (int)(unsigned)(st & 0xFFFFFFFFull);      // dim0_stride lo32
    g1[6] = (int)(unsigned)((st >> 32) & 0xFFFFull);  // dim0_stride hi16
    g1[7] = 0;

    i32x4 z4 = {0, 0, 0, 0};
#if defined(__clang_major__) && (__clang_major__ >= 23)
    i32x8 z8 = {0, 0, 0, 0, 0, 0, 0, 0};
    __builtin_amdgcn_tensor_load_to_lds(g0, g1, z4, z4, z8, 0);
#else
    __builtin_amdgcn_tensor_load_to_lds(g0, g1, z4, z4, 0);
#endif
}

__global__ __launch_bounds__(256)
void k_gemm_f16(const _Float16* __restrict__ A, const _Float16* __restrict__ W,
                const float* __restrict__ bias, const float* __restrict__ addsrc,
                float* __restrict__ C, int M, int Nn, int K, int act) {
    __shared__ _Float16 smem[4 * TILE_HALFS];  // A buf0/1, B buf0/1 (40 KB)

    const int tid  = threadIdx.x;
    const int lane = tid & 31;
    const int wave = tid >> 5;       // 0..7
    const int half = lane >> 4;      // lane group 0/1
    const int lrow = lane & 15;
    const int mg   = wave & 3;       // M group (32 rows each)
    const int ng   = wave >> 2;      // N group (64 cols each)

    const int m0 = blockIdx.y * BM;
    const int n0 = blockIdx.x * BN;

    const unsigned ldsBase = (unsigned)(unsigned long long)(const void*)smem;

    v8f acc[2][4] = {};

    const int nch = K / BK;

    // ---- prologue: prefetch chunks 0 and 1 via TDM (wave 0 only) ----
    if (wave == 0) {
        tdm_tile_load(ldsBase + 0u * TILE_HALFS * 2u, A,
                      (unsigned long long)(unsigned)m0 * (unsigned)K, K, M - m0);
        tdm_tile_load(ldsBase + 2u * TILE_HALFS * 2u, W,
                      (unsigned long long)(unsigned)n0 * (unsigned)K, K, Nn - n0);
        if (nch > 1) {
            tdm_tile_load(ldsBase + 1u * TILE_HALFS * 2u, A,
                          (unsigned long long)(unsigned)m0 * (unsigned)K + BK, K, M - m0);
            tdm_tile_load(ldsBase + 3u * TILE_HALFS * 2u, W,
                          (unsigned long long)(unsigned)n0 * (unsigned)K + BK, K, Nn - n0);
        }
    }

    for (int i = 0; i < nch; ++i) {
        if (wave == 0) {
            if (i + 1 < nch) __builtin_amdgcn_s_wait_tensorcnt((short)2);
            else             __builtin_amdgcn_s_wait_tensorcnt((short)0);
        }
        __syncthreads();   // chunk i resident; everyone past previous reads

        const _Float16* sA = smem + (unsigned)(i & 1) * TILE_HALFS;
        const _Float16* sB = smem + (unsigned)(2 + (i & 1)) * TILE_HALFS;

        // A 16x32 fragments (ISA 7.12.2): lane<16 K [0,8)+[16,24), lane>=16 +8
        v16h afr[2];
        #pragma unroll
        for (int mt = 0; mt < 2; ++mt) {
            const _Float16* ap = sA + (mg * 32 + mt * 16 + lrow) * LDST + half * 8;
            v8h lo = *(const v8h*)(ap);
            v8h hi = *(const v8h*)(ap + 16);
            afr[mt] = __builtin_shufflevector(lo, hi,
                      0,1,2,3,4,5,6,7,8,9,10,11,12,13,14,15);
        }
        // B 32x16 fragments: lane<16 K [0,8)+[16,24), lane>=16 +8 (same as A)
        v16h bfr[4];
        #pragma unroll
        for (int nt = 0; nt < 4; ++nt) {
            const _Float16* bp = sB + (ng * 64 + nt * 16 + lrow) * LDST + half * 8;
            v8h lo = *(const v8h*)(bp);
            v8h hi = *(const v8h*)(bp + 16);
            bfr[nt] = __builtin_shufflevector(lo, hi,
                      0,1,2,3,4,5,6,7,8,9,10,11,12,13,14,15);
        }

        #pragma unroll
        for (int mt = 0; mt < 2; ++mt)
            #pragma unroll
            for (int nt = 0; nt < 4; ++nt)
                acc[mt][nt] = __builtin_amdgcn_wmma_f32_16x16x32_f16(
                    false, afr[mt], false, bfr[nt], (short)0, acc[mt][nt],
                    false, false);

        __syncthreads();   // all reads of buf (i&1) done before TDM overwrites

        if (wave == 0 && i + 2 < nch) {
            unsigned long long k2 = (unsigned long long)(unsigned)((i + 2) * BK);
            tdm_tile_load(ldsBase + (unsigned)(i & 1) * TILE_HALFS * 2u, A,
                          (unsigned long long)(unsigned)m0 * (unsigned)K + k2,
                          K, M - m0);
            tdm_tile_load(ldsBase + (unsigned)(2 + (i & 1)) * TILE_HALFS * 2u, W,
                          (unsigned long long)(unsigned)n0 * (unsigned)K + k2,
                          K, Nn - n0);
        }
    }

    // C/D layout: lane<16 -> N=lane, M=+r ; lane>=16 -> N=lane-16, M=+r+8
    #pragma unroll
    for (int mt = 0; mt < 2; ++mt) {
        #pragma unroll
        for (int nt = 0; nt < 4; ++nt) {
            int n = n0 + ng * 64 + nt * 16 + lrow;
            if (n >= Nn) continue;
            #pragma unroll
            for (int r = 0; r < 8; ++r) {
                int m = m0 + mg * 32 + mt * 16 + r + half * 8;
                float v = acc[mt][nt][r];
                if (bias)     v += bias[n];
                if (act == 1) v = softplus_f(v);
                if (addsrc)   v += addsrc[(size_t)m * Nn + n];
                C[(size_t)m * Nn + n] = v;
            }
        }
    }
}

// ---------------- depthwise causal conv1d + SiLU ----------------
__global__ __launch_bounds__(256)
void k_conv_silu(const float* __restrict__ xi, const float* __restrict__ cw,
                 const float* __restrict__ cb, float* __restrict__ xact,
                 _Float16* __restrict__ xact16) {
    size_t idx = (size_t)blockIdx.x * blockDim.x + threadIdx.x;
    if (idx >= (size_t)MM * EE) return;
    int e   = (int)(idx % EE);
    int row = (int)(idx / EE);          // b*LL + l
    int l   = row % LL;
    int b   = row / LL;
    float acc = cb[e];
    #pragma unroll
    for (int j = 0; j < DCONV; ++j) {
        int ll = l - (DCONV - 1) + j;
        if (ll >= 0)
            acc += cw[e * DCONV + j] * xi[((size_t)(b * LL + ll)) * EE + e];
    }
    float v = silu_f(acc);
    xact[idx]   = v;
    xact16[idx] = (_Float16)v;
}

// ---------------- selective scan (sequential over L) ----------------
#define TCHUNK 128
__global__ __launch_bounds__(256)
void k_scan(const float* __restrict__ delta, const float* __restrict__ xact,
            const float* __restrict__ Bm, const float* __restrict__ Cm,
            const float* __restrict__ alog, float* __restrict__ y) {
    __shared__ float sB[TCHUNK * NN];
    __shared__ float sC[TCHUNK * NN];
    const int tid = threadIdx.x;
    const int e   = blockIdx.x * 256 + tid;
    const int b   = blockIdx.y;

    float Arow[NN];
    #pragma unroll
    for (int n = 0; n < NN; ++n) Arow[n] = -__expf(alog[(size_t)e * NN + n]);
    float h[NN];
    #pragma unroll
    for (int n = 0; n < NN; ++n) h[n] = 0.f;

    for (int t0 = 0; t0 < LL; t0 += TCHUNK) {
        __syncthreads();
        for (int i = tid; i < TCHUNK * NN; i += 256) {
            size_t g = (size_t)(b * LL + t0) * NN + i;
            sB[i] = Bm[g];
            sC[i] = Cm[g];
        }
        __syncthreads();
        for (int tt = 0; tt < TCHUNK; ++tt) {
            size_t g = (size_t)(b * LL + t0 + tt) * EE + e;
            float dt = delta[g];
            float xv = xact[g];
            float acc = 0.f;
            #pragma unroll
            for (int n = 0; n < NN; ++n) {
                float dA = __expf(dt * Arow[n]);
                h[n] = dA * h[n] + dt * sB[tt * NN + n] * xv;
                acc += h[n] * sC[tt * NN + n];
            }
            y[g] = acc;
        }
    }
}

// ---------------- gating: y = (y + x*wd) * silu(skip), to f16 ----------------
__global__ __launch_bounds__(256)
void k_ygate(const float* __restrict__ y, const float* __restrict__ xact,
             const float* __restrict__ wd, const float* __restrict__ skip,
             _Float16* __restrict__ y16) {
    size_t idx = (size_t)blockIdx.x * blockDim.x + threadIdx.x;
    if (idx >= (size_t)MM * EE) return;
    int e = (int)(idx % EE);
    float v = (y[idx] + xact[idx] * wd[e]) * silu_f(skip[idx]);
    y16[idx] = (_Float16)v;
}

// ---------------- host orchestration ----------------
static inline size_t alignUp256(size_t x) { return (x + 255) & ~(size_t)255; }

extern "C" void kernel_launch(void* const* d_in, const int* in_sizes, int n_in,
                              void* d_out, int out_size, void* d_ws, size_t ws_size,
                              hipStream_t stream) {
    (void)in_sizes; (void)n_in; (void)out_size; (void)ws_size;

    const int*   tokens    = (const int*)  d_in[0];
    const float* embedding = (const float*)d_in[1];
    const float* norm_w    = (const float*)d_in[2];
    const float* skip_w    = (const float*)d_in[3];
    const float* in_w      = (const float*)d_in[4];
    const float* conv_w    = (const float*)d_in[5];
    const float* conv_b    = (const float*)d_in[6];
    const float* Wd1       = (const float*)d_in[7];
    const float* Wd2_w     = (const float*)d_in[8];
    const float* Wd2_b     = (const float*)d_in[9];
    const float* WBp       = (const float*)d_in[10];
    const float* WCp       = (const float*)d_in[11];
    const float* A_log     = (const float*)d_in[12];
    const float* W_D       = (const float*)d_in[13];
    const float* out_w     = (const float*)d_in[14];
    const float* final_nw  = (const float*)d_in[15];
    const float* lm_head   = (const float*)d_in[16];
    float* out = (float*)d_out;

    // workspace carve-up
    char* ws = (char*)d_ws;
    size_t off = 0;
    auto alloc = [&](size_t bytes) -> char* {
        char* p = ws + off; off = alignUp256(off + bytes); return p;
    };
    float*    resid  = (float*)   alloc((size_t)MM * DD * 4);
    float*    resid2 = (float*)   alloc((size_t)MM * DD * 4);
    _Float16* xn16   = (_Float16*)alloc((size_t)MM * DD * 2);
    float*    skip   = (float*)   alloc((size_t)MM * EE * 4);
    float*    xi     = (float*)   alloc((size_t)MM * EE * 4);
    float*    xact   = (float*)   alloc((size_t)MM * EE * 4);
    _Float16* xact16 = (_Float16*)alloc((size_t)MM * EE * 2);
    float*    d1     = (float*)   alloc((size_t)MM * NDELTA * 4);
    _Float16* d116   = (_Float16*)alloc((size_t)MM * NDELTAP * 2);
    float*    delta  = (float*)   alloc((size_t)MM * EE * 4);
    float*    Bm     = (float*)   alloc((size_t)MM * NN * 4);
    float*    Cm     = (float*)   alloc((size_t)MM * NN * 4);
    float*    yv     = (float*)   alloc((size_t)MM * EE * 4);
    _Float16* y16    = (_Float16*)alloc((size_t)MM * EE * 2);
    _Float16* wbuf   = (_Float16*)alloc((size_t)VV * DD * 2); // largest weight

    auto cvt = [&](const float* src, _Float16* dst, int rows, int cols, int colsPad) {
        size_t total = (size_t)rows * colsPad;
        k_cvt_pad<<<dim3((unsigned)((total + 255) / 256)), 256, 0, stream>>>(
            src, dst, rows, cols, colsPad);
    };
    auto gemm = [&](const _Float16* A, const _Float16* W, const float* bias,
                    const float* addsrc, float* C, int M, int Nn, int K, int act) {
        dim3 grid((Nn + BN - 1) / BN, M / BM);
        k_gemm_f16<<<grid, 256, 0, stream>>>(A, W, bias, addsrc, C, M, Nn, K, act);
    };

    // 1) embedding gather
    {
        size_t total = (size_t)MM * DD;
        k_embed<<<dim3((unsigned)((total + 255) / 256)), 256, 0, stream>>>(
            tokens, embedding, resid);
    }

    const unsigned gME = (unsigned)(((size_t)MM * EE + 255) / 256);

    for (int i = 0; i < NLAYERS; ++i) {
        const float* nw   = norm_w + (size_t)i * DD;
        const float* skw  = skip_w + (size_t)i * EE * DD;
        const float* inw  = in_w   + (size_t)i * EE * DD;
        const float* cw   = conv_w + (size_t)i * EE * DCONV;
        const float* cb   = conv_b + (size_t)i * EE;
        const float* wd1  = Wd1    + (size_t)i * NDELTA * EE;
        const float* wd2w = Wd2_w  + (size_t)i * EE * NDELTA;
        const float* wd2b = Wd2_b  + (size_t)i * EE;
        const float* wb   = WBp    + (size_t)i * NN * EE;
        const float* wc   = WCp    + (size_t)i * NN * EE;
        const float* al   = A_log  + (size_t)i * EE * NN;
        const float* wd   = W_D    + (size_t)i * EE;
        const float* ow   = out_w  + (size_t)i * DD * EE;

        // x = rmsnorm(resid) -> f16
        k_rmsnorm<<<MM, 256, 0, stream>>>(resid, nw, xn16, DD);

        // skip = x @ skw.T ; xi = x @ inw.T
        cvt(skw, wbuf, EE, DD, DD);
        gemm(xn16, wbuf, nullptr, nullptr, skip, MM, EE, DD, 0);
        cvt(inw, wbuf, EE, DD, DD);
        gemm(xn16, wbuf, nullptr, nullptr, xi, MM, EE, DD, 0);

        // depthwise causal conv + SiLU
        k_conv_silu<<<gME, 256, 0, stream>>>(xi, cw, cb, xact, xact16);

        // delta = softplus((x @ wd1.T) @ wd2w.T + wd2b)
        cvt(wd1, wbuf, NDELTA, EE, EE);
        gemm(xact16, wbuf, nullptr, nullptr, d1, MM, NDELTA, EE, 0);
        cvt(d1, d116, MM, NDELTA, NDELTAP);              // pad K 48->64
        cvt(wd2w, wbuf, EE, NDELTA, NDELTAP);
        gemm(d116, wbuf, wd2b, nullptr, delta, MM, EE, NDELTAP, 1);

        // B/C projections
        cvt(wb, wbuf, NN, EE, EE);
        gemm(xact16, wbuf, nullptr, nullptr, Bm, MM, NN, EE, 0);
        cvt(wc, wbuf, NN, EE, EE);
        gemm(xact16, wbuf, nullptr, nullptr, Cm, MM, NN, EE, 0);

        // sequential selective scan
        k_scan<<<dim3(EE / 256, BB), 256, 0, stream>>>(delta, xact, Bm, Cm, al, yv);

        // gate, then resid2 = y @ ow.T + resid
        k_ygate<<<gME, 256, 0, stream>>>(yv, xact, wd, skip, y16);
        cvt(ow, wbuf, DD, EE, EE);
        gemm(y16, wbuf, nullptr, resid, resid2, MM, DD, EE, 0);

        // swap residual buffers
        float* t = resid; resid = resid2; resid2 = t;
    }

    // final norm + LM head
    k_rmsnorm<<<MM, 256, 0, stream>>>(resid, final_nw, xn16, DD);
    cvt(lm_head, wbuf, VV, DD, DD);
    gemm(xn16, wbuf, nullptr, nullptr, out, MM, VV, DD, 0);
}
